// RWKV_TimeMix_65369402245769
// MI455X (gfx1250) — matrix operand order repaired and checked by reading, not compiled, not run on hardware
//
#include <hip/hip_runtime.h>

typedef __attribute__((ext_vector_type(16))) _Float16 v16h;
typedef __attribute__((ext_vector_type(8)))  float    v8f;

#define B_DIM 16
#define T_DIM 768
#define C_DIM 768
#define M_DIM (B_DIM * T_DIM)   // 12288

// GEMM tiling
#define TM 128
#define TN 64
#define TK 32
#define LDSS 40   // padded LDS row stride in halves (80B -> conflict-free b128)

// ---------------------------------------------------------------------------
// 1) time-mix pre-pass: xk/xv/xr in f16
// ---------------------------------------------------------------------------
__global__ __launch_bounds__(256)
void mix_kernel(const float* __restrict__ x,
                const float* __restrict__ tmk,
                const float* __restrict__ tmv,
                const float* __restrict__ tmr,
                _Float16* __restrict__ xk,
                _Float16* __restrict__ xv,
                _Float16* __restrict__ xr) {
    int idx = blockIdx.x * blockDim.x + threadIdx.x;
    if (idx >= M_DIM * C_DIM) return;
    int c = idx % C_DIM;
    int t = (idx / C_DIM) % T_DIM;
    float xc = x[idx];
    float xp = (t == 0) ? 0.0f : x[idx - C_DIM];
    float mk = tmk[c], mv = tmv[c], mr = tmr[c];
    xk[idx] = (_Float16)(xc * mk + xp * (1.0f - mk));
    xv[idx] = (_Float16)(xc * mv + xp * (1.0f - mv));
    xr[idx] = (_Float16)(xc * mr + xp * (1.0f - mr));
}

// ---------------------------------------------------------------------------
// 2) weight f32 -> f16 conversion (all four weights in one pass)
// ---------------------------------------------------------------------------
__global__ __launch_bounds__(256)
void cvt_w(const float* __restrict__ a0, const float* __restrict__ a1,
           const float* __restrict__ a2, const float* __restrict__ a3,
           _Float16* __restrict__ o0, _Float16* __restrict__ o1,
           _Float16* __restrict__ o2, _Float16* __restrict__ o3) {
    int i = blockIdx.x * blockDim.x + threadIdx.x;
    if (i >= C_DIM * C_DIM) return;
    o0[i] = (_Float16)a0[i];
    o1[i] = (_Float16)a1[i];
    o2[i] = (_Float16)a2[i];
    o3[i] = (_Float16)a3[i];
}

// ---------------------------------------------------------------------------
// 3) Y[M,N] = A[M,K] * W[N,K]^T   (f16 in, f32 out, WMMA f32_16x16x32_f16)
//    block = 256 threads = 8 waves; wave grid 4x2; each wave: 32x32 output
// ---------------------------------------------------------------------------
__global__ __launch_bounds__(256)
void gemm_f16(const _Float16* __restrict__ A,   // M x K row-major
              const _Float16* __restrict__ W,   // N x K row-major
              float* __restrict__ Y,            // M x N row-major
              int M, int N, int K) {
    __shared__ __align__(16) _Float16 As[TM * LDSS];
    __shared__ __align__(16) _Float16 Bs[TN * LDSS];

    const int tid  = threadIdx.x;
    const int lane = tid & 31;
    const int wave = tid >> 5;     // 0..7
    const int wm   = wave >> 1;    // 0..3  (M direction)
    const int wn   = wave & 1;     // 0..1  (N direction)
    const int h    = lane >> 4;    // lane half (0/1)
    const int l    = lane & 15;

    const int m0 = blockIdx.y * TM;
    const int n0 = blockIdx.x * TN;

    // global->LDS staging indices
    const int arow = tid >> 1, aseg = tid & 1;   // 128 rows x 2 segs of 16 halves
    const int brow = tid >> 2, bseg = tid & 3;   // 64 rows x 4 segs of 8 halves

    v8f acc[2][2] = {};

    for (int k0 = 0; k0 < K; k0 += TK) {
        // stage A tile (128 x 32 halves)
        {
            const uint4* g = (const uint4*)(A + (size_t)(m0 + arow) * K + k0 + aseg * 16);
            uint4* s = (uint4*)(As + arow * LDSS + aseg * 16);
            s[0] = g[0];
            s[1] = g[1];
        }
        // stage W tile (64 x 32 halves)
        {
            const uint4* g = (const uint4*)(W + (size_t)(n0 + brow) * K + k0 + bseg * 8);
            uint4* s = (uint4*)(Bs + brow * LDSS + bseg * 8);
            s[0] = g[0];
        }
        if (k0 + TK < K) {
            __builtin_prefetch(A + (size_t)(m0 + arow) * K + k0 + TK + aseg * 16, 0, 0);
            __builtin_prefetch(W + (size_t)(n0 + brow) * K + k0 + TK + bseg * 8, 0, 0);
        }
        __syncthreads();

        // A fragment (16x32 f16): lanes 0-15 row m, halves e0-7 -> K=h*8+e,
        // e8-15 -> K=16+h*8+(e-8)   (ISA 7.12.2 16-bit A layout)
        // B fragment (32x16 f16): lane holds col n=l, halves e0-7 -> K=h*8+e, e8-15 -> K=16+h*8+(e-8)
        union FragU { v16h v; uint4 q[2]; };
        FragU a[2], b[2];
#pragma unroll
        for (int i = 0; i < 2; ++i) {
            int row = wm * 32 + i * 16 + l;
            const uint4* p0 = (const uint4*)(As + row * LDSS + h * 8);
            const uint4* p1 = (const uint4*)(As + row * LDSS + 16 + h * 8);
            a[i].q[0] = p0[0];
            a[i].q[1] = p1[0];
        }
#pragma unroll
        for (int j = 0; j < 2; ++j) {
            int col = wn * 32 + j * 16 + l;
            const uint4* p0 = (const uint4*)(Bs + col * LDSS + h * 8);
            const uint4* p1 = (const uint4*)(Bs + col * LDSS + 16 + h * 8);
            b[j].q[0] = p0[0];
            b[j].q[1] = p1[0];
        }

#pragma unroll
        for (int i = 0; i < 2; ++i)
#pragma unroll
            for (int j = 0; j < 2; ++j)
                acc[i][j] = __builtin_amdgcn_wmma_f32_16x16x32_f16(
                    false, a[i].v, false, b[j].v, (short)0, acc[i][j], false, false);

        __syncthreads();
    }

    // store: C/D layout — VGPR rr: M = rr + 8*h, N = l
#pragma unroll
    for (int i = 0; i < 2; ++i)
#pragma unroll
        for (int j = 0; j < 2; ++j) {
            int colg = n0 + wn * 32 + j * 16 + l;
#pragma unroll
            for (int rr = 0; rr < 8; ++rr) {
                int rowg = m0 + wm * 32 + i * 16 + h * 8 + rr;
                Y[(size_t)rowg * N + colg] = acc[i][j][rr];
            }
        }
}

// ---------------------------------------------------------------------------
// 4) WKV recurrence, fused exp / sigmoid / ratio; rwkv out in f16
//    one thread per (b,c); coalesced across c at each t
// ---------------------------------------------------------------------------
__global__ __launch_bounds__(256)
void wkv_kernel(const float* __restrict__ kraw,
                const float* __restrict__ vraw,
                const float* __restrict__ rraw,
                const float* __restrict__ time_decay,
                const float* __restrict__ time_first,
                _Float16* __restrict__ rwkv) {
    int id = blockIdx.x * blockDim.x + threadIdx.x;
    if (id >= B_DIM * C_DIM) return;
    int c = id % C_DIM;
    int b = id / C_DIM;

    float decay  = __expf(-__expf(time_decay[c]));  // per-step factor
    float wfirst = __expf(time_first[c]);           // "bonus" weight on current t

    float Aacc = 0.0f, Bacc = 0.0f;
    size_t base = (size_t)b * T_DIM * C_DIM + c;
    for (int t = 0; t < T_DIM; ++t) {
        size_t idx = base + (size_t)t * C_DIM;
        float kt = __expf(fminf(kraw[idx], 60.0f));
        float vt = vraw[idx];
        float rt = rraw[idx];
        float kv = kt * vt;
        float num = Aacc + wfirst * kv;
        float den = Bacc + wfirst * kt + 1e-8f;
        float sr  = 1.0f / (1.0f + __expf(-rt));
        rwkv[idx] = (_Float16)(sr * num / den);
        Aacc = Aacc * decay + kv;
        Bacc = Bacc * decay + kt;
    }
}

// ---------------------------------------------------------------------------
// launcher
// ---------------------------------------------------------------------------
extern "C" void kernel_launch(void* const* d_in, const int* in_sizes, int n_in,
                              void* d_out, int out_size, void* d_ws, size_t ws_size,
                              hipStream_t stream) {
    const float* x   = (const float*)d_in[0];
    const float* td  = (const float*)d_in[1];
    const float* tf  = (const float*)d_in[2];
    const float* tmk = (const float*)d_in[3];
    const float* tmv = (const float*)d_in[4];
    const float* tmr = (const float*)d_in[5];
    const float* Wk  = (const float*)d_in[6];
    const float* Wv  = (const float*)d_in[7];
    const float* Wr  = (const float*)d_in[8];
    const float* Wo  = (const float*)d_in[9];
    float* out = (float*)d_out;

    char* ws = (char*)d_ws;
    const size_t S16 = (size_t)M_DIM * C_DIM * sizeof(_Float16);  // 18.9 MB
    const size_t W16 = (size_t)C_DIM * C_DIM * sizeof(_Float16);  // 1.2 MB
    const size_t S32 = (size_t)M_DIM * C_DIM * sizeof(float);     // 37.7 MB

    _Float16* xk16 = (_Float16*)(ws);
    _Float16* xv16 = (_Float16*)(ws + S16);
    _Float16* xr16 = (_Float16*)(ws + 2 * S16);
    _Float16* wk16 = (_Float16*)(ws + 3 * S16);
    _Float16* wv16 = (_Float16*)(ws + 3 * S16 + W16);
    _Float16* wr16 = (_Float16*)(ws + 3 * S16 + 2 * W16);
    _Float16* wo16 = (_Float16*)(ws + 3 * S16 + 3 * W16);
    float*    kraw = (float*)   (ws + 3 * S16 + 4 * W16);
    float*    vraw = (float*)   (ws + 3 * S16 + 4 * W16 + S32);
    float*    rraw = (float*)   (ws + 3 * S16 + 4 * W16 + 2 * S32);
    _Float16* rwkv = (_Float16*)(ws + 3 * S16 + 4 * W16 + 3 * S32);

    // 1) mixes -> f16
    {
        int n = M_DIM * C_DIM;
        mix_kernel<<<(n + 255) / 256, 256, 0, stream>>>(x, tmk, tmv, tmr,
                                                        xk16, xv16, xr16);
    }
    // 2) weights -> f16
    {
        int n = C_DIM * C_DIM;
        cvt_w<<<(n + 255) / 256, 256, 0, stream>>>(Wk, Wv, Wr, Wo,
                                                   wk16, wv16, wr16, wo16);
    }
    // 3) k/v/r projections
    dim3 grid(C_DIM / TN, M_DIM / TM);   // (12, 96)
    gemm_f16<<<grid, 256, 0, stream>>>(xk16, wk16, kraw, M_DIM, C_DIM, C_DIM);
    gemm_f16<<<grid, 256, 0, stream>>>(xv16, wv16, vraw, M_DIM, C_DIM, C_DIM);
    gemm_f16<<<grid, 256, 0, stream>>>(xr16, wr16, rraw, M_DIM, C_DIM, C_DIM);

    // 4) WKV recurrence + gating -> rwkv (f16)
    {
        int n = B_DIM * C_DIM;
        wkv_kernel<<<(n + 255) / 256, 256, 0, stream>>>(kraw, vraw, rraw,
                                                        td, tf, rwkv);
    }
    // 5) output projection -> d_out (f32, (b,t,o))
    gemm_f16<<<grid, 256, 0, stream>>>(rwkv, wo16, out, M_DIM, C_DIM, C_DIM);
}
